// AttentionLayer_257698038341
// MI455X (gfx1250) — matrix operand order repaired and checked by reading, not compiled, not run on hardware
//
#include <hip/hip_runtime.h>
#include <math.h>

#define LSEQ 2048
#define BB   4
#define DD   1024
#define HH   16
#define DKK  64
#define DFF_ 4096

typedef __attribute__((ext_vector_type(16))) _Float16 v16h;
typedef __attribute__((ext_vector_type(8)))  _Float16 h8;
typedef __attribute__((ext_vector_type(4)))  _Float16 h4;
typedef __attribute__((ext_vector_type(8)))  float    v8f;
typedef __attribute__((ext_vector_type(4)))  float    f4;

union Frag16 { v16h v; h8 h[2]; };

__device__ __forceinline__ v8f wmma16(v16h a, v16h b, v8f c) {
  return __builtin_amdgcn_wmma_f32_16x16x32_f16(false, a, false, b, (short)0, c,
                                                false, false);
}

// gfx1250 async copy: global -> LDS, 16 bytes per lane, tracked by ASYNCcnt.
__device__ __forceinline__ void async_cp16(unsigned lds_off, const void* gptr) {
  unsigned long long ga = (unsigned long long)(size_t)gptr;
  asm volatile("global_load_async_to_lds_b128 %0, %1, off"
               :: "v"(lds_off), "v"(ga) : "memory");
}
__device__ __forceinline__ void wait_async0() {
  asm volatile("s_wait_asynccnt 0x0" ::: "memory");
}

// ---------------------------------------------------------------------------
// fp32 -> fp16 elementwise (vectorized x4)
// ---------------------------------------------------------------------------
__global__ __launch_bounds__(256)
void cvt_f16(const float* __restrict__ in, _Float16* __restrict__ out, int n4) {
  int i = blockIdx.x * 256 + threadIdx.x;
  if (i < n4) {
    f4 a = *(const f4*)(in + (size_t)i * 4);
    h4 h = {(_Float16)a.x, (_Float16)a.y, (_Float16)a.z, (_Float16)a.w};
    *(h4*)(out + (size_t)i * 4) = h;
  }
}

// ---------------------------------------------------------------------------
// W[K,N] fp32 -> Wt[N,K] fp16 (LDS tile transpose, 32x32 per block)
// grid (N/32, K/32)
// ---------------------------------------------------------------------------
__global__ __launch_bounds__(256)
void cvt_transpose(const float* __restrict__ W, _Float16* __restrict__ Wt,
                   int K, int N) {
  __shared__ _Float16 tl[32][33];
  const int n0 = blockIdx.x * 32, k0 = blockIdx.y * 32;
  const int c = threadIdx.x & 31, r = threadIdx.x >> 5;  // r: 0..7
#pragma unroll
  for (int p = 0; p < 4; p++) {
    int k = r + p * 8;
    tl[c][k] = (_Float16)W[(size_t)(k0 + k) * N + n0 + c];
  }
  __syncthreads();
#pragma unroll
  for (int p = 0; p < 4; p++) {
    int n = r + p * 8;
    Wt[(size_t)(n0 + n) * K + k0 + c] = tl[n][c];
  }
}

// ---------------------------------------------------------------------------
// GEMM: C[M,N] = A[M,K](f16) @ Bt[N,K](f16, pre-transposed) + bias, * sc
// EPI=1 -> exact GELU. OUTF16 selects f16/f32 output.
// Block tile 128x128, K-step 32, double-buffered LDS filled by
// global_load_async_to_lds_b128; compute overlaps the async DMA.
// ---------------------------------------------------------------------------
template <int EPI, int OUTF16>
__global__ __launch_bounds__(256)
void gemm_f16w(const _Float16* __restrict__ A, const _Float16* __restrict__ Bt,
               const float* __restrict__ bias, void* __restrict__ Cout,
               float sc, int M, int N, int K) {
  __shared__ _Float16 As[2][128][40];   // [m][k], 40-pad: conflict-free, 16B rows
  __shared__ _Float16 Bs[2][128][40];   // [n][k]
  const int tid = threadIdx.x;
  const int wave = tid >> 5, lane = tid & 31;
  const int cl = lane & 15, hb = lane >> 4;
  const int wm = wave >> 2, wn = wave & 3;
  const int bn0 = blockIdx.x * 128, bm0 = blockIdx.y * 128;
  (void)M;

  v8f acc[4][2];
#pragma unroll
  for (int i = 0; i < 4; i++)
#pragma unroll
    for (int j = 0; j < 2; j++) acc[i][j] = {};

  const int nk = K >> 5;
  const int rA = tid >> 2, seg = tid & 3;  // 4 x 16B segments per 32-half row

  // prologue: async-stage tile 0 into buffer 0
#pragma unroll
  for (int p = 0; p < 2; p++) {
    int rr = rA + p * 64;
    async_cp16((unsigned)(size_t)&As[0][rr][seg * 8],
               A + (size_t)(bm0 + rr) * K + seg * 8);
    async_cp16((unsigned)(size_t)&Bs[0][rr][seg * 8],
               Bt + (size_t)(bn0 + rr) * K + seg * 8);
  }

  for (int kb = 0; kb < nk; ++kb) {
    const int buf = kb & 1;
    wait_async0();        // this wave's copies for tile kb are in LDS
    __syncthreads();      // everyone's copies done; prev compute on buf^1 done
    if (kb + 1 < nk) {    // overlap: DMA tile kb+1 while computing tile kb
#pragma unroll
      for (int p = 0; p < 2; p++) {
        int rr = rA + p * 64;
        async_cp16((unsigned)(size_t)&As[buf ^ 1][rr][seg * 8],
                   A + (size_t)(bm0 + rr) * K + (kb + 1) * 32 + seg * 8);
        async_cp16((unsigned)(size_t)&Bs[buf ^ 1][rr][seg * 8],
                   Bt + (size_t)(bn0 + rr) * K + (kb + 1) * 32 + seg * 8);
      }
    }

    Frag16 af[4], bf[2];
#pragma unroll
    for (int mt = 0; mt < 4; ++mt) {  // A frag: K = hb*16..+15
      int row = wm * 64 + mt * 16 + cl;
      af[mt].h[0] = *(const h8*)&As[buf][row][hb * 16];
      af[mt].h[1] = *(const h8*)&As[buf][row][hb * 16 + 8];
    }
#pragma unroll
    for (int nt = 0; nt < 2; ++nt) {  // B frag: K = hb*16..+15 of col cl
      int col = wn * 32 + nt * 16 + cl;
      bf[nt].h[0] = *(const h8*)&Bs[buf][col][hb * 16];
      bf[nt].h[1] = *(const h8*)&Bs[buf][col][hb * 16 + 8];
    }
#pragma unroll
    for (int mt = 0; mt < 4; ++mt)
#pragma unroll
      for (int nt = 0; nt < 2; ++nt)
        acc[mt][nt] = wmma16(af[mt].v, bf[nt].v, acc[mt][nt]);
  }

  // epilogue: acc[..][i] -> row = i + 8*hb, col = cl (per 16x16 tile)
#pragma unroll
  for (int nt = 0; nt < 2; ++nt) {
    int cn = bn0 + wn * 32 + nt * 16 + cl;
    float bv = bias[cn];
#pragma unroll
    for (int mt = 0; mt < 4; ++mt) {
#pragma unroll
      for (int i = 0; i < 8; i++) {
        int rm = bm0 + wm * 64 + mt * 16 + i + 8 * hb;
        float v = (acc[mt][nt][i] + bv) * sc;
        if (EPI == 1) v = 0.5f * v * (1.0f + erff(v * 0.70710678118654752f));
        if (OUTF16)
          ((_Float16*)Cout)[(size_t)rm * N + cn] = (_Float16)v;
        else
          ((float*)Cout)[(size_t)rm * N + cn] = v;
      }
    }
  }
}

// ---------------------------------------------------------------------------
// Flash attention over f16 q/k/v (scale prefolded into q). grid (B*H, L/128).
// 8 waves; each wave owns 16 q rows x full 64-key tile (softmax stats reduce
// only across 16 lanes). K tile staged by async DMA; V transposed via VALU.
// ---------------------------------------------------------------------------
__global__ __launch_bounds__(256)
void flash_attn(const _Float16* __restrict__ Q, const _Float16* __restrict__ Kg,
                const _Float16* __restrict__ Vg, _Float16* __restrict__ O) {
  __shared__ _Float16 Ks[64][72];     // [key][dk]
  __shared__ _Float16 Vt[64][72];     // [dk][key]
  __shared__ _Float16 Ps[8][16][72];  // per-wave P tile [qrow][key]
  const int tid = threadIdx.x;
  const int wave = tid >> 5, lane = tid & 31;
  const int cl = lane & 15, hb = lane >> 4;
  const int b = blockIdx.x >> 4, h = blockIdx.x & 15;
  const int qt = blockIdx.y;

  // Q fragments live in VGPRs for the whole kernel (already f16 + pre-scaled)
  const int lq = qt * 128 + wave * 16 + cl;
  const _Float16* qrow = Q + ((size_t)lq * BB + b) * DD + h * DKK;
  Frag16 qf[2];
#pragma unroll
  for (int c = 0; c < 2; c++) {
    qf[c].h[0] = *(const h8*)(qrow + c * 32 + hb * 16);
    qf[c].h[1] = *(const h8*)(qrow + c * 32 + hb * 16 + 8);
  }

  float m_i[8], l_i[8];
#pragma unroll
  for (int i = 0; i < 8; i++) { m_i[i] = -INFINITY; l_i[i] = 0.0f; }
  v8f oacc[4];
#pragma unroll
  for (int ot = 0; ot < 4; ot++) oacc[ot] = {};

  for (int kt = 0; kt < LSEQ / 64; ++kt) {
    __syncthreads();  // previous iteration's reads of Ks/Vt complete
    // K tile: pure async DMA (row-major == B-frag layout for Q@K^T)
#pragma unroll
    for (int p = 0; p < 2; p++) {
      int e = p * 256 + tid;
      int kr = e >> 3, sg = e & 7;
      async_cp16((unsigned)(size_t)&Ks[kr][sg * 8],
                 Kg + ((size_t)(kt * 64 + kr) * BB + b) * DD + h * DKK + sg * 8);
    }
    // V tile: load f16 rows, transpose-store into Vt[dk][key]
#pragma unroll
    for (int p = 0; p < 2; p++) {
      int e = p * 256 + tid;
      int kr = e >> 3, c8 = (e & 7) * 8;
      h8 vv = *(const h8*)(Vg + ((size_t)(kt * 64 + kr) * BB + b) * DD + h * DKK + c8);
#pragma unroll
      for (int j = 0; j < 8; j++) Vt[c8 + j][kr] = vv[j];
    }
    wait_async0();
    __syncthreads();

    // S = Q @ K^T (16 rows x 64 keys per wave; K-dim 64 = 2 WMMA steps)
    v8f s[4];
#pragma unroll
    for (int nt = 0; nt < 4; ++nt) {
      s[nt] = {};
#pragma unroll
      for (int c = 0; c < 2; c++) {
        Frag16 kf;
        kf.h[0] = *(const h8*)&Ks[nt * 16 + cl][c * 32 + hb * 16];
        kf.h[1] = *(const h8*)&Ks[nt * 16 + cl][c * 32 + hb * 16 + 8];
        s[nt] = wmma16(qf[c].v, kf.v, s[nt]);
      }
    }

    // online softmax; row = i + 8*hb, cols spread across 16 lanes x 4 tiles
#pragma unroll
    for (int i = 0; i < 8; i++) {
      float tm = fmaxf(fmaxf(s[0][i], s[1][i]), fmaxf(s[2][i], s[3][i]));
#pragma unroll
      for (int mk = 1; mk < 16; mk <<= 1) tm = fmaxf(tm, __shfl_xor(tm, mk, 32));
      float mn = fmaxf(m_i[i], tm);
      float sum = 0.0f;
#pragma unroll
      for (int nt = 0; nt < 4; nt++) {
        float p = __expf(s[nt][i] - mn);
        s[nt][i] = p;
        sum += p;
      }
#pragma unroll
      for (int mk = 1; mk < 16; mk <<= 1) sum += __shfl_xor(sum, mk, 32);
      float alpha = __expf(m_i[i] - mn);
      l_i[i] = l_i[i] * alpha + sum;
      m_i[i] = mn;
#pragma unroll
      for (int ot = 0; ot < 4; ot++) oacc[ot][i] = oacc[ot][i] * alpha;
#pragma unroll
      for (int nt = 0; nt < 4; nt++)
        Ps[wave][i + 8 * hb][nt * 16 + cl] = (_Float16)s[nt][i];
    }
    asm volatile("" ::: "memory");  // keep Ps stores before Ps reads (DS in-order)

    // O += P @ V (K-dim 64 keys = 2 steps; 4 dk tiles)
#pragma unroll
    for (int kc = 0; kc < 2; kc++) {
      Frag16 pf;
      pf.h[0] = *(const h8*)&Ps[wave][cl][kc * 32 + hb * 16];
      pf.h[1] = *(const h8*)&Ps[wave][cl][kc * 32 + hb * 16 + 8];
#pragma unroll
      for (int ot = 0; ot < 4; ot++) {
        Frag16 vf;
        vf.h[0] = *(const h8*)&Vt[ot * 16 + cl][kc * 32 + hb * 16];
        vf.h[1] = *(const h8*)&Vt[ot * 16 + cl][kc * 32 + hb * 16 + 8];
        oacc[ot] = wmma16(pf.v, vf.v, oacc[ot]);
      }
    }
  }

  // normalize by l and write f16 context (L*B, D)
#pragma unroll
  for (int ot = 0; ot < 4; ot++) {
#pragma unroll
    for (int i = 0; i < 8; i++) {
      int row = qt * 128 + wave * 16 + i + 8 * hb;
      float inv = 1.0f / l_i[i];
      O[((size_t)row * BB + b) * DD + h * DKK + ot * 16 + cl] =
          (_Float16)(oacc[ot][i] * inv);
    }
  }
}

// ---------------------------------------------------------------------------
// Fused residual-add + LayerNorm. Writes f32 result and optional f16 copy.
// ---------------------------------------------------------------------------
__global__ __launch_bounds__(256)
void add_ln(const float* __restrict__ X, const float* __restrict__ Y,
            const float* __restrict__ g, const float* __restrict__ be,
            float* __restrict__ out, _Float16* __restrict__ outh) {
  const int row = blockIdx.x, t = threadIdx.x;
  const size_t base = (size_t)row * DD + t * 4;
  f4 a = *(const f4*)(X + base);
  f4 y = *(const f4*)(Y + base);
  f4 sv = a + y;
  float s = sv.x + sv.y + sv.z + sv.w;
  float ss = sv.x * sv.x + sv.y * sv.y + sv.z * sv.z + sv.w * sv.w;
#pragma unroll
  for (int mk = 1; mk < 32; mk <<= 1) {
    s += __shfl_xor(s, mk, 32);
    ss += __shfl_xor(ss, mk, 32);
  }
  __shared__ float rs[8], rss[8], smu, srstd;
  int wv = t >> 5, ln = t & 31;
  if (ln == 0) { rs[wv] = s; rss[wv] = ss; }
  __syncthreads();
  if (t == 0) {
    float S = 0.0f, SS = 0.0f;
    for (int w = 0; w < 8; w++) { S += rs[w]; SS += rss[w]; }
    float mu = S * (1.0f / DD);
    float var = SS * (1.0f / DD) - mu * mu;
    smu = mu;
    srstd = rsqrtf(var + 1e-5f);
  }
  __syncthreads();
  f4 gv = *(const f4*)(g + t * 4);
  f4 bv = *(const f4*)(be + t * 4);
  f4 o = (sv - smu) * srstd * gv + bv;
  *(f4*)(out + base) = o;
  if (outh) {
    h4 oh = {(_Float16)o.x, (_Float16)o.y, (_Float16)o.z, (_Float16)o.w};
    *(h4*)(outh + base) = oh;
  }
}

// ---------------------------------------------------------------------------
extern "C" void kernel_launch(void* const* d_in, const int* in_sizes, int n_in,
                              void* d_out, int out_size, void* d_ws, size_t ws_size,
                              hipStream_t stream) {
  (void)in_sizes; (void)n_in; (void)out_size; (void)ws_size;
  const float* x   = (const float*)d_in[0];
  const float* Wq  = (const float*)d_in[1];
  const float* bq  = (const float*)d_in[2];
  const float* Wk  = (const float*)d_in[3];
  const float* bk  = (const float*)d_in[4];
  const float* Wv  = (const float*)d_in[5];
  const float* bv  = (const float*)d_in[6];
  const float* Wo  = (const float*)d_in[7];
  const float* bo  = (const float*)d_in[8];
  const float* W1  = (const float*)d_in[9];
  const float* b1  = (const float*)d_in[10];
  const float* W2  = (const float*)d_in[11];
  const float* b2  = (const float*)d_in[12];
  const float* g1  = (const float*)d_in[13];
  const float* be1 = (const float*)d_in[14];
  const float* g2  = (const float*)d_in[15];
  const float* be2 = (const float*)d_in[16];

  const int M = LSEQ * BB;                 // 8192 rows
  const size_t nx = (size_t)M * DD;        // 8.39M elements
  char* base = (char*)d_ws;
  // f16 activations
  _Float16* qh   = (_Float16*)(base + 0 * nx);          // 2*nx bytes
  _Float16* kh   = (_Float16*)(base + 2 * nx);
  _Float16* vh   = (_Float16*)(base + 4 * nx);
  _Float16* ctxh = (_Float16*)(base + 6 * nx);
  _Float16* hbuf = (_Float16*)(base + 0 * nx);          // reuse [0,8nx): M*DFF f16
  _Float16* xh   = (_Float16*)(base + 8 * nx);          // 2*nx bytes
  // f16 transposed weights
  _Float16* wqt = (_Float16*)(base + 10 * nx);                       // 2 MiB
  _Float16* wkt = (_Float16*)(base + 10 * nx + (2u << 20));
  _Float16* wvt = (_Float16*)(base + 10 * nx + (4u << 20));
  _Float16* wot = (_Float16*)(base + 10 * nx + (6u << 20));
  _Float16* w1t = (_Float16*)(base + 10 * nx + (8u << 20));          // 8 MiB
  _Float16* w2t = (_Float16*)(base + 10 * nx + (16u << 20));         // 8 MiB
  // f32 buffers
  float* proj = (float*)(base + 10 * nx + (24u << 20));              // 4*nx bytes
  float* ff   = proj;                                                // reuse
  float* x1   = (float*)(base + 14 * nx + (24u << 20));              // 4*nx bytes
  _Float16* x1h = (_Float16*)(base + 18 * nx + (24u << 20));         // 2*nx bytes

  dim3 blk(256);
  // one-time conversions (per launch): x -> f16, weights -> f16 transposed
  cvt_f16<<<(int)(nx / 1024), blk, 0, stream>>>(x, xh, (int)(nx / 4));
  cvt_transpose<<<dim3(DD / 32, DD / 32), blk, 0, stream>>>(Wq, wqt, DD, DD);
  cvt_transpose<<<dim3(DD / 32, DD / 32), blk, 0, stream>>>(Wk, wkt, DD, DD);
  cvt_transpose<<<dim3(DD / 32, DD / 32), blk, 0, stream>>>(Wv, wvt, DD, DD);
  cvt_transpose<<<dim3(DD / 32, DD / 32), blk, 0, stream>>>(Wo, wot, DD, DD);
  cvt_transpose<<<dim3(DFF_ / 32, DD / 32), blk, 0, stream>>>(W1, w1t, DD, DFF_);
  cvt_transpose<<<dim3(DD / 32, DFF_ / 32), blk, 0, stream>>>(W2, w2t, DFF_, DD);

  // QKV projections (q gets 1/sqrt(dk) folded in)
  gemm_f16w<0, 1><<<dim3(DD / 128, M / 128), blk, 0, stream>>>(xh, wqt, bq, qh, 0.125f, M, DD, DD);
  gemm_f16w<0, 1><<<dim3(DD / 128, M / 128), blk, 0, stream>>>(xh, wkt, bk, kh, 1.0f, M, DD, DD);
  gemm_f16w<0, 1><<<dim3(DD / 128, M / 128), blk, 0, stream>>>(xh, wvt, bv, vh, 1.0f, M, DD, DD);
  flash_attn<<<dim3(BB * HH, LSEQ / 128), blk, 0, stream>>>(qh, kh, vh, ctxh);
  gemm_f16w<0, 0><<<dim3(DD / 128, M / 128), blk, 0, stream>>>(ctxh, wot, bo, proj, 1.0f, M, DD, DD);
  add_ln<<<M, blk, 0, stream>>>(x, proj, g1, be1, x1, x1h);
  gemm_f16w<1, 1><<<dim3(DFF_ / 128, M / 128), blk, 0, stream>>>(x1h, w1t, b1, hbuf, 1.0f, M, DFF_, DD);
  gemm_f16w<0, 0><<<dim3(DD / 128, M / 128), blk, 0, stream>>>(hbuf, w2t, b2, ff, 1.0f, M, DD, DFF_);
  add_ln<<<M, blk, 0, stream>>>(x1, ff, g2, be2, (float*)d_out, (_Float16*)nullptr);
}
